// Seq2Seq_2267742732349
// MI455X (gfx1250) — matrix operand order repaired and checked by reading, not compiled, not run on hardware
//
#include <hip/hip_runtime.h>
#include <hip/hip_bf16.h>
#include <math.h>

// ---------------------------------------------------------------------------
// Seq2Seq GRU (enc 64 steps, dec 47 steps) + per-step NLL.
// bf16 WMMA GEMMs, async global->LDS staging (CDNA5), double-buffered LDS.
// ---------------------------------------------------------------------------

typedef unsigned short u16;
typedef __attribute__((ext_vector_type(16))) __bf16 v16bf;
typedef __attribute__((ext_vector_type(8)))  __bf16 v8bf;
typedef __attribute__((ext_vector_type(8)))  float  v8f;

#define HH   1024   // hidden
#define EE   512    // embed
#define BB   2048   // batch
#define G3   3072   // 3*H
#define S_ENC 64
#define S_DEC 47    // last decoder step's label is ignore_index -> skip it
#define NK   (HH / 32)

static __device__ __forceinline__ u16 f32_to_bf16(float f) {
    union { float f; unsigned int u; } x; x.f = f;
    unsigned int u = x.u;
    u += 0x7FFFu + ((u >> 16) & 1u);     // round-to-nearest-even
    return (u16)(u >> 16);
}

// CDNA5 async copy: 16B per lane, global -> LDS, tracked by ASYNCcnt.
// Low 32 bits of a flat shared-memory pointer are the LDS offset (ISA 10.2).
static __device__ __forceinline__ void async_ld_b128(void* lds_dst, const void* gsrc) {
    unsigned lds_off = (unsigned)(unsigned long long)lds_dst;
    asm volatile("global_load_async_to_lds_b128 %0, %1, off"
                 :: "v"(lds_off), "v"(gsrc)
                 : "memory");
}
static __device__ __forceinline__ void wait_async0() {
    asm volatile("s_wait_asynccnt 0x0" ::: "memory");
}

// ---------------------------------------------------------------- setup ----
__global__ void init_state_kernel(float* h0f, u16* h0b, float* acc, int n) {
    int i = blockIdx.x * blockDim.x + threadIdx.x;
    if (i < n) { h0f[i] = 0.0f; h0b[i] = 0; }
    if (i == 0) acc[0] = 0.0f;
}

__global__ void cvt_bf16_kernel(u16* dst, const float* src, int n) {
    int i = blockIdx.x * blockDim.x + threadIdx.x;
    if (i < n) dst[i] = f32_to_bf16(src[i]);
}

// pad unembedding to 256 classes: rows 254/255 -> W=0, bias=-1e30 (mask cols)
__global__ void build_unemb_kernel(u16* wpad, float* bpad,
                                   const float* W, const float* b) {
    int i = blockIdx.x * blockDim.x + threadIdx.x;   // 256*1024 threads
    int v = i >> 10, k = i & 1023;
    wpad[i] = (v < 254) ? f32_to_bf16(W[v * HH + k]) : (u16)0;
    if (i < 256) bpad[i] = (i < 254) ? b[i] : -1.0e30f;
}

// tab[v,g] = bih[g] + sum_e emb[v,e]*Wih[g,e]  (fp32 exact; 256x3072)
__global__ void build_tab_kernel(float* tab, const float* emb,
                                 const float* Wih, const float* bih) {
    int i = blockIdx.x * blockDim.x + threadIdx.x;   // 256*3072 threads
    int v = i / G3, g = i % G3;
    const float* e = emb + v * EE;
    const float* w = Wih + g * EE;
    float acc = bih[g];
    #pragma unroll 8
    for (int k = 0; k < EE; ++k) acc = fmaf(e[k], w[k], acc);
    tab[i] = acc;
}

// ------------------------------------------------------------- GRU step ----
// Fused gh = h @ Whh^T (r,z,n together) + gate math + state update.
// Block tile 128x64; 8 waves, each 16 rows x 64 cols x 3 gates.
// Double-buffered LDS, async global->LDS staging overlapped with WMMA.
__launch_bounds__(256)
__global__ void gru_step_kernel(const u16*  __restrict__ hin_bf,
                                const float*__restrict__ hin_f,
                                u16*        __restrict__ hout_bf,
                                float*      __restrict__ hout_f,
                                const u16*  __restrict__ whh_bf,   // [3072][1024]
                                const float*__restrict__ bhh,      // [3072]
                                const float*__restrict__ tab,      // [256][3072]
                                const int*  __restrict__ seq,      // [S][2048]
                                int step) {
    __shared__ u16 lds_a[2 * 128 * 32];   // h tile, [row][k]
    __shared__ u16 lds_b[2 * 192 * 32];   // Whh tile, [gate*64 + n][k]

    const int tid  = threadIdx.x;
    const int rowBase = blockIdx.x * 128;
    const int colBase = blockIdx.y * 64;
    const int wave = tid >> 5;
    const int lane = tid & 31;
    const int nl   = lane & 15;       // N / A-row within frag
    const int hi   = lane >> 4;       // half-wave select

    v8f acc[3][4];
    #pragma unroll
    for (int g = 0; g < 3; ++g)
        #pragma unroll
        for (int s = 0; s < 4; ++s)
            #pragma unroll
            for (int j = 0; j < 8; ++j) acc[g][s][j] = 0.0f;

    auto stage = [&](int kc, int p) {
        const int kBase = kc * 32;
        u16* la = lds_a + p * (128 * 32);
        u16* lb = lds_b + p * (192 * 32);
        // A: 128x32 bf16 = 512 x 16B, 2 per thread (async, no VGPR data)
        #pragma unroll
        for (int it = 0; it < 2; ++it) {
            int idx = tid + it * 256;
            int r = idx >> 2, seg = idx & 3;
            async_ld_b128(la + r * 32 + seg * 8,
                          hin_bf + (size_t)(rowBase + r) * HH + kBase + seg * 8);
        }
        // B: 192x32 bf16 = 768 x 16B, 3 per thread
        #pragma unroll
        for (int it = 0; it < 3; ++it) {
            int idx = tid + it * 256;
            int rr = idx >> 2, seg = idx & 3;
            int g = rr >> 6, n = rr & 63;
            async_ld_b128(lb + rr * 32 + seg * 8,
                          whh_bf + (size_t)(g * HH + colBase + n) * HH + kBase + seg * 8);
        }
    };

    stage(0, 0);
    wait_async0();
    __syncthreads();

    for (int kc = 0; kc < NK; ++kc) {
        const int p = kc & 1;
        if (kc + 1 < NK) {
            stage(kc + 1, p ^ 1);          // async fill of idle buffer
            __builtin_prefetch(hin_bf + (size_t)(rowBase + (tid >> 1)) * HH + kc * 32 + 64, 0, 3);
        }

        const u16* la = lds_a + p * (128 * 32);
        const u16* lb = lds_b + p * (192 * 32);

        // A fragment (16x32 bf16): lane row = nl; contiguous K = hi*16 + j
        const u16* ap = la + (wave * 16 + nl) * 32;
        v8bf alo = *(const v8bf*)(ap + hi * 16);
        v8bf ahi = *(const v8bf*)(ap + hi * 16 + 8);
        v16bf af;
        #pragma unroll
        for (int j = 0; j < 8; ++j) { af[j] = alo[j]; af[j + 8] = ahi[j]; }

        // 3 gates x 4 N-subtiles; batch the 4 B-frag loads ahead of the WMMAs
        #pragma unroll
        for (int g = 0; g < 3; ++g) {
            v16bf bf[4];
            #pragma unroll
            for (int sub = 0; sub < 4; ++sub)
                bf[sub] = *(const v16bf*)(lb + (g * 64 + sub * 16 + nl) * 32 + hi * 16);
            #pragma unroll
            for (int sub = 0; sub < 4; ++sub)
                acc[g][sub] = __builtin_amdgcn_wmma_f32_16x16x32_bf16(
                    false, af, false, bf[sub], (short)0, acc[g][sub], false, false);
        }

        wait_async0();     // next chunk's async stores have landed
        __syncthreads();
    }

    // --- epilogue: gates + state update (C/D layout: m = hi*8 + i, n = nl)
    int toks[8];
    #pragma unroll
    for (int i = 0; i < 8; ++i)
        toks[i] = seq[step * BB + rowBase + wave * 16 + hi * 8 + i];

    #pragma unroll
    for (int sub = 0; sub < 4; ++sub) {
        int col = colBase + sub * 16 + nl;
        float br = bhh[col], bz = bhh[HH + col], bn = bhh[2 * HH + col];
        #pragma unroll
        for (int i = 0; i < 8; ++i) {
            int row = rowBase + wave * 16 + hi * 8 + i;
            const float* t = tab + (size_t)toks[i] * G3;
            float xr = t[col], xz = t[HH + col], xn = t[2 * HH + col];
            float gr = acc[0][sub][i] + br;
            float gz = acc[1][sub][i] + bz;
            float gn = acc[2][sub][i] + bn;
            float r = 1.0f / (1.0f + __expf(-(xr + gr)));
            float z = 1.0f / (1.0f + __expf(-(xz + gz)));
            float n = tanhf(xn + r * gn);
            float hprev = hin_f[(size_t)row * HH + col];
            float hnew  = (1.0f - z) * n + z * hprev;
            hout_f[(size_t)row * HH + col]  = hnew;
            hout_bf[(size_t)row * HH + col] = f32_to_bf16(hnew);
        }
    }
}

// ------------------------------------------------------------ loss step ----
// logits[32 rows][256 cls] = h_bf @ WuPad^T + biasPad, then row log-softmax NLL.
__launch_bounds__(256)
__global__ void loss_step_kernel(const u16*  __restrict__ h_bf,
                                 const u16*  __restrict__ wpad,   // [256][1024]
                                 const float*__restrict__ bpad,   // [256]
                                 const int*  __restrict__ tgt,    // [48][2048]
                                 int step,
                                 float*      __restrict__ acc_out) {
    __shared__ __align__(16) char smem[36864];   // 36 KB
    u16*   lds_a = (u16*)smem;                   // 2 x 32x32  (4 KB)
    u16*   lds_b = (u16*)(smem + 4096);          // 2 x 256x32 (32 KB)
    float* lds_l = (float*)smem;                 // 32x256 fp32, overlay after GEMM
    __shared__ float sred;

    const int tid  = threadIdx.x;
    const int rowBase = blockIdx.x * 32;
    const int wave = tid >> 5, lane = tid & 31;
    const int wm = wave >> 2, wn = wave & 3;     // 2 waves in M x 4 in N
    const int nl = lane & 15, hi = lane >> 4;

    v8f acc[4];
    #pragma unroll
    for (int s = 0; s < 4; ++s)
        #pragma unroll
        for (int j = 0; j < 8; ++j) acc[s][j] = 0.0f;

    auto stage = [&](int kc, int p) {
        const int kBase = kc * 32;
        u16* la = lds_a + p * (32 * 32);
        u16* lb = lds_b + p * (256 * 32);
        if (tid < 128) {                         // A: 32x32 = 128 x 16B
            int r = tid >> 2, seg = tid & 3;
            async_ld_b128(la + r * 32 + seg * 8,
                          h_bf + (size_t)(rowBase + r) * HH + kBase + seg * 8);
        }
        #pragma unroll
        for (int it = 0; it < 4; ++it) {         // B: 256x32 = 1024 x 16B
            int idx = tid + it * 256;
            int n = idx >> 2, seg = idx & 3;
            async_ld_b128(lb + n * 32 + seg * 8,
                          wpad + (size_t)n * HH + kBase + seg * 8);
        }
    };

    stage(0, 0);
    wait_async0();
    __syncthreads();

    for (int kc = 0; kc < NK; ++kc) {
        const int p = kc & 1;
        if (kc + 1 < NK) stage(kc + 1, p ^ 1);

        const u16* la = lds_a + p * (32 * 32);
        const u16* lb = lds_b + p * (256 * 32);

        const u16* ap = la + (wm * 16 + nl) * 32;
        v8bf alo = *(const v8bf*)(ap + hi * 16);
        v8bf ahi = *(const v8bf*)(ap + hi * 16 + 8);
        v16bf af;
        #pragma unroll
        for (int j = 0; j < 8; ++j) { af[j] = alo[j]; af[j + 8] = ahi[j]; }

        v16bf bf[4];
        #pragma unroll
        for (int sub = 0; sub < 4; ++sub)
            bf[sub] = *(const v16bf*)(lb + (wn * 64 + sub * 16 + nl) * 32 + hi * 16);
        #pragma unroll
        for (int sub = 0; sub < 4; ++sub)
            acc[sub] = __builtin_amdgcn_wmma_f32_16x16x32_bf16(
                false, af, false, bf[sub], (short)0, acc[sub], false, false);

        wait_async0();
        __syncthreads();
    }

    // tiles dead; overlay smem with logits
    #pragma unroll
    for (int sub = 0; sub < 4; ++sub) {
        int col = wn * 64 + sub * 16 + nl;
        float b = bpad[col];
        #pragma unroll
        for (int i = 0; i < 8; ++i) {
            int rl = wm * 16 + hi * 8 + i;
            lds_l[rl * 256 + col] = acc[sub][i] + b;
        }
    }
    if (tid == 0) sred = 0.0f;
    __syncthreads();

    if (tid < 32) {                 // one thread per row: log-softmax NLL
        const float* L = lds_l + tid * 256;
        float m = -3.0e38f;
        for (int n = 0; n < 256; ++n) m = fmaxf(m, L[n]);
        float s = 0.0f;
        for (int n = 0; n < 256; ++n) s += __expf(L[n] - m);
        int label = tgt[(step + 1) * BB + rowBase + tid];   // in [2,256)
        float nll = __logf(s) + m - L[label - 2];
        atomicAdd(&sred, nll);      // ds_add_f32
    }
    __syncthreads();
    if (tid == 0) atomicAdd(acc_out, sred);   // global_atomic_add_f32
}

__global__ void finalize_kernel(float* out, const float* acc) {
    if (blockIdx.x == 0 && threadIdx.x == 0)
        out[0] = acc[0] / (float)(S_DEC * BB);
}

// ------------------------------------------------------------- launcher ----
extern "C" void kernel_launch(void* const* d_in, const int* in_sizes, int n_in,
                              void* d_out, int out_size, void* d_ws, size_t ws_size,
                              hipStream_t stream) {
    const int*   src_seq = (const int*)  d_in[0];
    const int*   tgt_seq = (const int*)  d_in[1];
    const float* enc_emb = (const float*)d_in[2];
    const float* enc_Wih = (const float*)d_in[3];
    const float* enc_Whh = (const float*)d_in[4];
    const float* enc_bih = (const float*)d_in[5];
    const float* enc_bhh = (const float*)d_in[6];
    const float* dec_emb = (const float*)d_in[7];
    const float* dec_Wih = (const float*)d_in[8];
    const float* dec_Whh = (const float*)d_in[9];
    const float* dec_bih = (const float*)d_in[10];
    const float* dec_bhh = (const float*)d_in[11];
    const float* unemb_W = (const float*)d_in[12];
    const float* unemb_b = (const float*)d_in[13];

    char* ws = (char*)d_ws;
    size_t off = 0;
    auto alloc = [&](size_t bytes) -> char* {
        char* p = ws + off;
        off = (off + bytes + 255) & ~(size_t)255;
        return p;
    };
    float* encTab  = (float*)alloc((size_t)256 * G3 * 4);     // 3 MB
    float* decTab  = (float*)alloc((size_t)256 * G3 * 4);     // 3 MB
    u16*   encWhhB = (u16*)  alloc((size_t)G3 * HH * 2);      // 6 MB
    u16*   decWhhB = (u16*)  alloc((size_t)G3 * HH * 2);      // 6 MB
    u16*   wpad    = (u16*)  alloc((size_t)256 * HH * 2);     // 0.5 MB
    float* bpad    = (float*)alloc(256 * 4);
    float* hf[2]; u16* hb[2];
    hf[0] = (float*)alloc((size_t)BB * HH * 4);
    hf[1] = (float*)alloc((size_t)BB * HH * 4);
    hb[0] = (u16*)  alloc((size_t)BB * HH * 2);
    hb[1] = (u16*)  alloc((size_t)BB * HH * 2);
    float* accum = (float*)alloc(256);
    // total ~44.6 MB of d_ws

    // ---- per-call setup (deterministic; rebuilt every launch)
    init_state_kernel<<<(BB * HH + 255) / 256, 256, 0, stream>>>(hf[0], hb[0], accum, BB * HH);
    cvt_bf16_kernel<<<(G3 * HH + 255) / 256, 256, 0, stream>>>(encWhhB, enc_Whh, G3 * HH);
    cvt_bf16_kernel<<<(G3 * HH + 255) / 256, 256, 0, stream>>>(decWhhB, dec_Whh, G3 * HH);
    build_unemb_kernel<<<(256 * HH) / 256, 256, 0, stream>>>(wpad, bpad, unemb_W, unemb_b);
    build_tab_kernel<<<(256 * G3) / 256, 256, 0, stream>>>(encTab, enc_emb, enc_Wih, enc_bih);
    build_tab_kernel<<<(256 * G3) / 256, 256, 0, stream>>>(decTab, dec_emb, dec_Wih, dec_bih);

    dim3 ggrid(BB / 128, HH / 64);   // 16 x 16 blocks
    int cur = 0;

    // ---- encoder: 64 steps
    for (int t = 0; t < S_ENC; ++t) {
        gru_step_kernel<<<ggrid, 256, 0, stream>>>(
            hb[cur], hf[cur], hb[1 - cur], hf[1 - cur],
            encWhhB, enc_bhh, encTab, src_seq, t);
        cur ^= 1;
    }
    // ---- decoder: 47 steps, each followed by its NLL contribution
    for (int s = 0; s < S_DEC; ++s) {
        gru_step_kernel<<<ggrid, 256, 0, stream>>>(
            hb[cur], hf[cur], hb[1 - cur], hf[1 - cur],
            decWhhB, dec_bhh, decTab, tgt_seq, s);
        cur ^= 1;
        loss_step_kernel<<<BB / 32, 256, 0, stream>>>(
            hb[cur], wpad, bpad, tgt_seq, s, accum);
    }

    finalize_kernel<<<1, 64, 0, stream>>>((float*)d_out, accum);
}
